// DisentangledSelfAttentionV2_49795850830371
// MI455X (gfx1250) — matrix operand order repaired and checked by reading, not compiled, not run on hardware
//
#include <hip/hip_runtime.h>
#include <hip/hip_bf16.h>
#include <math.h>

// ---- problem constants (from reference) ----
#define B_      2
#define L_      1024
#define D_      1024
#define H_      16
#define HEAD_   64
#define SPAN_   256          // ATT_SPAN
#define P2_     512          // 2*ATT_SPAN
#define BUCKETS_ 256
#define MAXREL_ 512

typedef __bf16 bf16;
typedef __attribute__((ext_vector_type(16))) __bf16 v16bf;
typedef __attribute__((ext_vector_type(8)))  float  v8f;

__device__ __forceinline__ v8f wmma_bf16(v16bf a, v16bf b, v8f c) {
    // D = A(16x32 bf16) x B(32x16 bf16) + C(16x16 f32)
    return __builtin_amdgcn_wmma_f32_16x16x32_bf16(
        /*neg_a=*/false, a, /*neg_b=*/false, b,
        /*c_mod=*/(short)0, c, /*reuse_a=*/false, /*reuse_b=*/false);
}

// ---- fragment loaders (ISA 7.12.2 layouts, wave32) ----
// A matrix 16x32 (16-bit): lane m = lane&15, half = lane>>4
//   VGPR v (v<4): K = half*8 + 2v + {0,1}; VGPR v (v>=4): K = 16 + half*8 + 2(v-4) + {0,1}
// -> per lane: 8 contiguous bf16 at row[hf*8] and 8 at row[16 + hf*8] (vectorizes to 2x b128)
__device__ __forceinline__ v16bf load_a_bf16(const bf16* __restrict__ src, int ld) {
    int lane = threadIdx.x & 31;
    int m = lane & 15, hf = lane >> 4;
    const bf16* row = src + (size_t)m * ld;
    v16bf a;
#pragma unroll
    for (int v = 0; v < 8; ++v) {
        int kb = ((v < 4) ? 0 : 16) + hf * 8 + (v & 3) * 2;
        a[2 * v]     = row[kb];
        a[2 * v + 1] = row[kb + 1];
    }
    return a;
}

// B matrix 32x16 (16-bit): lane n = lane&15, half = lane>>4; same K order as A: K = half*8 + 0..7, then 16 + half*8 + 0..7
// element (k,n) at src[k*strideK + n*strideN]
__device__ __forceinline__ v16bf load_b_bf16(const bf16* __restrict__ src, int strideK, int strideN) {
    int lane = threadIdx.x & 31;
    int n = lane & 15;
    int kg = (lane >> 4) * 8;
    const bf16* col = src + (size_t)n * strideN;
    v16bf b;
#pragma unroll
    for (int v = 0; v < 8; ++v) {
        int k = ((v < 4) ? 0 : 16) + kg + (v & 3) * 2;
        b[2 * v]     = col[(size_t)k * strideK];
        b[2 * v + 1] = col[(size_t)(k + 1) * strideK];
    }
    return b;
}

// C/D 16x16 f32: lane n = lane&15, half = lane>>4; VGPR r holds row m = r + 8*half
__device__ __forceinline__ void store_c_f32(float* __restrict__ dst, int ld, v8f c) {
    int lane = threadIdx.x & 31;
    int n = lane & 15, hf = lane >> 4;
#pragma unroll
    for (int r = 0; r < 8; ++r) dst[(size_t)(r + 8 * hf) * ld + n] = c[r];
}

// ---- fp32 -> bf16 convert ----
__global__ void k_f2bf(const float* __restrict__ in, bf16* __restrict__ out, int n) {
    int i = blockIdx.x * blockDim.x + threadIdx.x;
    for (; i < n; i += gridDim.x * blockDim.x) out[i] = (bf16)in[i];
}

// ---- log-bucket (matches make_log_bucket_position) ----
__device__ __forceinline__ int log_bucket(int rel) {
    const int mid = BUCKETS_ / 2; // 128
    int ar = rel < 0 ? -rel : rel;
    float abs_pos = (ar < mid) ? (float)(mid - 1) : (float)ar;
    if (abs_pos <= (float)mid) return rel;
    float log_pos = ceilf(logf(abs_pos / (float)mid) /
                          logf((float)(MAXREL_ - 1) / (float)mid) * (float)(mid - 1)) + (float)mid;
    float sgn = (rel > 0) ? 1.0f : ((rel < 0) ? -1.0f : 0.0f);
    return (int)(log_pos * sgn);
}

// ---- bucket LUT: idx[d] for rel = d - (L_-1), d in [0, 2L-2] ----
// p2c_idx[k,q] = clip(-bucket(k-q)+S) = clip(bucket(q-k)+S) == c2p idx (odd symmetry),
// so a single LUT serves both gathers.
__global__ void k_bucket_lut(int* __restrict__ lut) {
    int i = blockIdx.x * blockDim.x + threadIdx.x;
    if (i < 2 * L_ - 1) {
        int rel = i - (L_ - 1);
        int idx = log_bucket(rel) + SPAN_;
        idx = idx < 0 ? 0 : (idx > P2_ - 1 ? P2_ - 1 : idx);
        lut[i] = idx;
    }
}

// ---- projection GEMM: out[i,j] = sum_k X[i,k]*W[j,k] + bias[j]  (out bf16 [M,D]) ----
// one wave computes a 16x64 strip (4 accumulators share one A fragment); grid = (M/16, D/64)
__global__ void k_proj(const bf16* __restrict__ X, const bf16* __restrict__ W,
                       const float* __restrict__ bias, bf16* __restrict__ out) {
    int m0 = blockIdx.x * 16, n0 = blockIdx.y * 64;
    v8f c0 = {}, c1 = {}, c2 = {}, c3 = {};
    for (int k0 = 0; k0 < D_; k0 += 32) {
        const bf16* abase = X + (size_t)m0 * D_ + k0;
        if (k0 + 32 < D_) {
            int lane = threadIdx.x & 31;
            __builtin_prefetch(abase + (size_t)(lane & 15) * D_ + 32, 0, 3);
        }
        v16bf a = load_a_bf16(abase, D_);
        v16bf b0 = load_b_bf16(W + (size_t)(n0 +  0) * D_ + k0, 1, D_);
        v16bf b1 = load_b_bf16(W + (size_t)(n0 + 16) * D_ + k0, 1, D_);
        v16bf b2 = load_b_bf16(W + (size_t)(n0 + 32) * D_ + k0, 1, D_);
        v16bf b3 = load_b_bf16(W + (size_t)(n0 + 48) * D_ + k0, 1, D_);
        c0 = wmma_bf16(a, b0, c0);
        c1 = wmma_bf16(a, b1, c1);
        c2 = wmma_bf16(a, b2, c2);
        c3 = wmma_bf16(a, b3, c3);
    }
    int lane = threadIdx.x & 31;
    int n = lane & 15, hf = lane >> 4;
    v8f acc[4] = {c0, c1, c2, c3};
#pragma unroll
    for (int j = 0; j < 4; ++j) {
        float bj = bias[n0 + j * 16 + n];
#pragma unroll
        for (int r = 0; r < 8; ++r)
            out[(size_t)(m0 + r + 8 * hf) * D_ + n0 + j * 16 + n] = (bf16)(acc[j][r] + bj);
    }
}

// ---- position attention: out[bh][i,p] = sum_d QK[b,i,h*64+d] * POS[p,h*64+d] ----
// one wave computes 16 x 64 strip of the [L, 512] table; grid = (L/16, P2/64, B*H)
__global__ void k_posatt(const bf16* __restrict__ QK, const bf16* __restrict__ POS,
                         float* __restrict__ out) {
    int i0 = blockIdx.x * 16, p0 = blockIdx.y * 64, bh = blockIdx.z;
    int b = bh / H_, h = bh % H_;
    const bf16* abase = QK + ((size_t)b * L_ + i0) * D_ + h * HEAD_;
    v8f c0 = {}, c1 = {}, c2 = {}, c3 = {};
#pragma unroll
    for (int k0 = 0; k0 < HEAD_; k0 += 32) {
        v16bf a = load_a_bf16(abase + k0, D_);
        const bf16* bb = POS + h * HEAD_ + k0;
        v16bf b0 = load_b_bf16(bb + (size_t)(p0 +  0) * D_, 1, D_);
        v16bf b1 = load_b_bf16(bb + (size_t)(p0 + 16) * D_, 1, D_);
        v16bf b2 = load_b_bf16(bb + (size_t)(p0 + 32) * D_, 1, D_);
        v16bf b3 = load_b_bf16(bb + (size_t)(p0 + 48) * D_, 1, D_);
        c0 = wmma_bf16(a, b0, c0);
        c1 = wmma_bf16(a, b1, c1);
        c2 = wmma_bf16(a, b2, c2);
        c3 = wmma_bf16(a, b3, c3);
    }
    float* obase = out + ((size_t)bh * L_ + i0) * P2_ + p0;
    store_c_f32(obase +  0, P2_, c0);
    store_c_f32(obase + 16, P2_, c1);
    store_c_f32(obase + 32, P2_, c2);
    store_c_f32(obase + 48, P2_, c3);
}

// ---- scores: QK^T + gathered c2p + p2c, scaled + mask bias ----
// one wave computes a 16 x 64 strip of scores; grid = (L/16, L/64, B*H)
__global__ void k_scores(const bf16* __restrict__ Qb, const bf16* __restrict__ Kb,
                         const float* __restrict__ c2p, const float* __restrict__ p2c,
                         const int* __restrict__ mask, const int* __restrict__ lut,
                         float* __restrict__ scores) {
    int q0 = blockIdx.x * 16, kbase = blockIdx.y * 64, bh = blockIdx.z;
    int b = bh / H_, h = bh % H_;
    const bf16* abase = Qb + ((size_t)b * L_ + q0) * D_ + h * HEAD_;
    v8f c0 = {}, c1 = {}, c2 = {}, c3 = {};
#pragma unroll
    for (int kk = 0; kk < HEAD_; kk += 32) {
        v16bf a = load_a_bf16(abase + kk, D_);
        const bf16* bb = Kb + ((size_t)b * L_ + kbase) * D_ + h * HEAD_ + kk;
        v16bf b0 = load_b_bf16(bb + (size_t) 0 * D_, 1, D_);
        v16bf b1 = load_b_bf16(bb + (size_t)16 * D_, 1, D_);
        v16bf b2 = load_b_bf16(bb + (size_t)32 * D_, 1, D_);
        v16bf b3 = load_b_bf16(bb + (size_t)48 * D_, 1, D_);
        c0 = wmma_bf16(a, b0, c0);
        c1 = wmma_bf16(a, b1, c1);
        c2 = wmma_bf16(a, b2, c2);
        c3 = wmma_bf16(a, b3, c3);
    }
    const float inv_scale = 0.07216878364870322f; // 1/sqrt(64*3)
    int lane = threadIdx.x & 31;
    int n = lane & 15, hf = lane >> 4;
    const float* c2p_bh = c2p + (size_t)bh * L_ * P2_;
    v8f acc[4] = {c0, c1, c2, c3};
#pragma unroll
    for (int j = 0; j < 4; ++j) {
        int kcol = kbase + j * 16 + n;
        float mbias = (mask[b * L_ + kcol] > 0) ? 0.0f : -1e9f;
        const float* p2c_row = p2c + ((size_t)bh * L_ + kcol) * P2_;
        const int* lrow = lut + (L_ - 1) - kcol;   // lrow[qrow] = idx(qrow - kcol)
#pragma unroll
        for (int r = 0; r < 8; ++r) {
            int qrow = q0 + r + 8 * hf;
            int idx = lrow[qrow];
            float val = acc[j][r] + c2p_bh[(size_t)qrow * P2_ + idx] + p2c_row[idx];
            scores[((size_t)bh * L_ + qrow) * L_ + kcol] = val * inv_scale + mbias;
        }
    }
}

// ---- row softmax in place, row length L_ ----
__global__ void k_softmax(float* __restrict__ scores) {
    __shared__ float red[256];
    float* p = scores + (size_t)blockIdx.x * L_;
    int t = threadIdx.x;
    float m = -1e30f;
    for (int i = t; i < L_; i += 256) m = fmaxf(m, p[i]);
    red[t] = m; __syncthreads();
    for (int s = 128; s > 0; s >>= 1) { if (t < s) red[t] = fmaxf(red[t], red[t + s]); __syncthreads(); }
    m = red[0]; __syncthreads();
    float sum = 0.0f;
    for (int i = t; i < L_; i += 256) { float e = expf(p[i] - m); p[i] = e; sum += e; }
    red[t] = sum; __syncthreads();
    for (int s = 128; s > 0; s >>= 1) { if (t < s) red[t] += red[t + s]; __syncthreads(); }
    float inv = 1.0f / red[0];
    for (int i = t; i < L_; i += 256) p[i] *= inv;
}

// ---- context: out[b,q,h*64+n] = sum_k probs[bh,q,k] * V[b,k,h*64+n] ----
// one wave computes the full 16 x 64 head strip; grid = (L/16, B*H)
__global__ void k_context(const float* __restrict__ probs, const bf16* __restrict__ Vb,
                          float* __restrict__ out) {
    int q0 = blockIdx.x * 16, bh = blockIdx.y;
    int b = bh / H_, h = bh % H_;
    int lane = threadIdx.x & 31;
    int m = lane & 15, hf = lane >> 4;
    v8f c0 = {}, c1 = {}, c2 = {}, c3 = {};
    for (int k0 = 0; k0 < L_; k0 += 32) {
        // A fragment: probs (fp32) converted to bf16 on the fly
        const float* prow = probs + ((size_t)bh * L_ + q0 + m) * L_ + k0;
        if (k0 + 32 < L_) __builtin_prefetch(prow + 32, 0, 3);
        v16bf a;
#pragma unroll
        for (int v = 0; v < 8; ++v) {
            int kb = ((v < 4) ? 0 : 16) + hf * 8 + (v & 3) * 2;
            a[2 * v]     = (bf16)prow[kb];
            a[2 * v + 1] = (bf16)prow[kb + 1];
        }
        const bf16* bb = Vb + ((size_t)b * L_ + k0) * D_ + h * HEAD_;
        v16bf b0 = load_b_bf16(bb +  0, D_, 1);
        v16bf b1 = load_b_bf16(bb + 16, D_, 1);
        v16bf b2 = load_b_bf16(bb + 32, D_, 1);
        v16bf b3 = load_b_bf16(bb + 48, D_, 1);
        c0 = wmma_bf16(a, b0, c0);
        c1 = wmma_bf16(a, b1, c1);
        c2 = wmma_bf16(a, b2, c2);
        c3 = wmma_bf16(a, b3, c3);
    }
    int n = lane & 15;
    v8f acc[4] = {c0, c1, c2, c3};
#pragma unroll
    for (int j = 0; j < 4; ++j)
#pragma unroll
        for (int r = 0; r < 8; ++r) {
            int q = q0 + r + 8 * hf;
            out[((size_t)b * L_ + q) * D_ + h * HEAD_ + j * 16 + n] = acc[j][r];
        }
}

// ================= host side =================
extern "C" void kernel_launch(void* const* d_in, const int* in_sizes, int n_in,
                              void* d_out, int out_size, void* d_ws, size_t ws_size,
                              hipStream_t stream) {
    const float* X   = (const float*)d_in[0];   // [B,L,D]
    const int*   msk = (const int*)d_in[1];     // [B,L]
    const float* RE  = (const float*)d_in[2];   // [2S, D]
    const float* Wq  = (const float*)d_in[3];
    const float* bq  = (const float*)d_in[4];
    const float* Wk  = (const float*)d_in[5];
    const float* bk  = (const float*)d_in[6];
    const float* Wv  = (const float*)d_in[7];
    const float* bv  = (const float*)d_in[8];
    float* out = (float*)d_out;

    // workspace carve-up
    char* p = (char*)d_ws;
    auto take = [&](size_t bytes) { char* r = p; p += (bytes + 255) & ~(size_t)255; return r; };
    bf16* Xbf   = (bf16*)take((size_t)B_ * L_ * D_ * 2);
    bf16* Wqb   = (bf16*)take((size_t)D_ * D_ * 2);
    bf16* Wkb   = (bf16*)take((size_t)D_ * D_ * 2);
    bf16* Wvb   = (bf16*)take((size_t)D_ * D_ * 2);
    bf16* REb   = (bf16*)take((size_t)P2_ * D_ * 2);
    bf16* Qb    = (bf16*)take((size_t)B_ * L_ * D_ * 2);
    bf16* Kb    = (bf16*)take((size_t)B_ * L_ * D_ * 2);
    bf16* Vb    = (bf16*)take((size_t)B_ * L_ * D_ * 2);
    bf16* PQb   = (bf16*)take((size_t)P2_ * D_ * 2);
    bf16* PKb   = (bf16*)take((size_t)P2_ * D_ * 2);
    float* c2p  = (float*)take((size_t)B_ * H_ * L_ * P2_ * 4);
    float* p2c  = (float*)take((size_t)B_ * H_ * L_ * P2_ * 4);
    float* sc   = (float*)take((size_t)B_ * H_ * L_ * L_ * 4);
    int*   lut  = (int*)take((size_t)(2 * L_ - 1) * 4);

    // 1) converts + bucket LUT
    {
        int n1 = B_ * L_ * D_, n2 = D_ * D_, n3 = P2_ * D_;
        k_f2bf<<<(n1 + 255) / 256, 256, 0, stream>>>(X, Xbf, n1);
        k_f2bf<<<(n2 + 255) / 256, 256, 0, stream>>>(Wq, Wqb, n2);
        k_f2bf<<<(n2 + 255) / 256, 256, 0, stream>>>(Wk, Wkb, n2);
        k_f2bf<<<(n2 + 255) / 256, 256, 0, stream>>>(Wv, Wvb, n2);
        k_f2bf<<<(n3 + 255) / 256, 256, 0, stream>>>(RE, REb, n3);
        k_bucket_lut<<<(2 * L_ - 1 + 255) / 256, 256, 0, stream>>>(lut);
    }

    dim3 wave(32);

    // 2) projections (Q,K,V) and position projections (pos_k uses Wk, pos_q uses Wq)
    {
        dim3 g((B_ * L_) / 16, D_ / 64);
        k_proj<<<g, wave, 0, stream>>>(Xbf, Wqb, bq, Qb);
        k_proj<<<g, wave, 0, stream>>>(Xbf, Wkb, bk, Kb);
        k_proj<<<g, wave, 0, stream>>>(Xbf, Wvb, bv, Vb);
        dim3 gp(P2_ / 16, D_ / 64);
        k_proj<<<gp, wave, 0, stream>>>(REb, Wkb, bk, PKb);
        k_proj<<<gp, wave, 0, stream>>>(REb, Wqb, bq, PQb);
    }

    // 3) c2p_att = Q . pos_k^T ; p2c_att = K . pos_q^T
    {
        dim3 g(L_ / 16, P2_ / 64, B_ * H_);
        k_posatt<<<g, wave, 0, stream>>>(Qb, PKb, c2p);
        k_posatt<<<g, wave, 0, stream>>>(Kb, PQb, p2c);
    }

    // 4) scores = (QK^T + c2p_gather + p2c_gather)/sqrt(192) + mask bias
    {
        dim3 g(L_ / 16, L_ / 64, B_ * H_);
        k_scores<<<g, wave, 0, stream>>>(Qb, Kb, c2p, p2c, msk, lut, sc);
    }

    // 5) softmax per row
    k_softmax<<<B_ * H_ * L_, 256, 0, stream>>>(sc);

    // 6) context = probs @ V, written as [B, L, D]
    {
        dim3 g(L_ / 16, B_ * H_);
        k_context<<<g, wave, 0, stream>>>(sc, Vb, out);
    }
    (void)in_sizes; (void)n_in; (void)out_size; (void)ws_size;
}
